// Net1_73847667687728
// MI455X (gfx1250) — matrix operand order repaired and checked by reading, not compiled, not run on hardware
//
#include <hip/hip_runtime.h>

// ---------- raw clang vector types (trivial, union-safe) ----------
typedef __attribute__((ext_vector_type(16))) __bf16    v16bf;
typedef __attribute__((ext_vector_type(8)))  float     v8f;
typedef __attribute__((ext_vector_type(4)))  float     f32x4;
typedef __attribute__((ext_vector_type(4)))  unsigned  u32x4;

union FragU {
    u32x4 u[2];
    v16bf v;
    unsigned short s[16];
};

__device__ __forceinline__ unsigned short f2bf(float f) {
    unsigned u = __float_as_uint(f);
    unsigned r = u + 0x7FFFu + ((u >> 16) & 1u);   // round-to-nearest-even
    return (unsigned short)(r >> 16);
}
__device__ __forceinline__ float bf2f(unsigned short h) {
    return __uint_as_float(((unsigned)h) << 16);
}
__device__ __forceinline__ float celu13(float x) {
    return x > 0.f ? x : 1.3f * (expf(x * (1.f / 1.3f)) - 1.f);
}

// CDNA5 async DMA: global -> LDS, 16B per lane, ASYNCcnt-tracked.
// INST_OFFSET is added to BOTH the LDS and the global address (ISA 08 §4.4),
// which matches our tiles (64 contiguous bytes per row on both sides).
__device__ __forceinline__ void async_b128(unsigned lds_byte, const void* gaddr, int ioff) {
    asm volatile("global_load_async_to_lds_b128 %0, %1, off offset:%2"
                 :: "v"(lds_byte), "v"(gaddr), "i"(ioff) : "memory");
}
__device__ __forceinline__ void wait_async0() {
    asm volatile("s_wait_asynccnt 0" ::: "memory");
}

// ---------------- constants ----------------
#define BATCH 32
#define MLEN  2048
#define EDIM  1024
#define HEADS 8
#define HD    128
#define MID   64
#define TM    64     // rows per workgroup
#define TN    256    // cols per workgroup (= 2 heads)
#define TK    32
#define KPAD  40     // padded K-stride (ushorts): 80B rows, 16B aligned
#define CPAD  264    // padded C stride (ushorts): 528B rows

// ============================================================
// transpose fp32 [R][C] -> bf16 [C][R]
// ============================================================
__global__ __launch_bounds__(256)
void transpose_to_bf16(const float* __restrict__ in, unsigned short* __restrict__ out,
                       int R, int C) {
    __shared__ float tile[32][33];
    int c0 = blockIdx.x * 32, r0 = blockIdx.y * 32;
    int tx = threadIdx.x, ty = threadIdx.y;          // 32 x 8
    #pragma unroll
    for (int i = 0; i < 32; i += 8)
        tile[ty + i][tx] = in[(r0 + ty + i) * C + c0 + tx];
    __syncthreads();
    #pragma unroll
    for (int i = 0; i < 32; i += 8)
        out[(c0 + ty + i) * R + r0 + tx] = f2bf(tile[tx][ty + i]);
}

__global__ void zero_kernel(float* p, int n) {
    int i = blockIdx.x * blockDim.x + threadIdx.x;
    if (i < n) p[i] = 0.f;
}

// ============================================================
// q projections: [32,1024] @ [1024,1024] + bias -> CELU -> GroupNorm(8)
// ============================================================
__global__ __launch_bounds__(256)
void qproj_kernel(const float* __restrict__ query, const float* __restrict__ W,
                  const float* __restrict__ bvec, const float* __restrict__ gwv,
                  const float* __restrict__ gbv, float* __restrict__ qout) {
    __shared__ float wt[32 * 128];
    __shared__ float qt[32 * 32];
    __shared__ float cbuf[32 * 128];
    const int g = blockIdx.x;
    const int t = threadIdx.x;
    const int c = t & 127, rg = t >> 7;

    float accv[16];
    #pragma unroll
    for (int i = 0; i < 16; ++i) accv[i] = 0.f;

    for (int kb = 0; kb < 32; ++kb) {
        __syncthreads();
        {   // W tile [32k][128c]
            int li = t * 16, kr = li >> 7, cc = li & 127;
            const f32x4* src = (const f32x4*)(W + (kb * 32 + kr) * EDIM + g * 128 + cc);
            f32x4* dst = (f32x4*)(wt + kr * 128 + cc);
            dst[0] = src[0]; dst[1] = src[1]; dst[2] = src[2]; dst[3] = src[3];
        }
        {   // query tile [32r][32k]
            int li = t * 4, r = li >> 5, kk = li & 31;
            *(f32x4*)(qt + r * 32 + kk) = *(const f32x4*)(query + r * EDIM + kb * 32 + kk);
        }
        __syncthreads();
        #pragma unroll
        for (int i = 0; i < 16; ++i) {
            int row = rg * 16 + i;
            float a = 0.f;
            #pragma unroll
            for (int kk = 0; kk < 32; ++kk)
                a = fmaf(qt[row * 32 + kk], wt[kk * 128 + c], a);
            accv[i] += a;
        }
    }
    __syncthreads();
    float cb = bvec[g * 128 + c];
    #pragma unroll
    for (int i = 0; i < 16; ++i)
        cbuf[(rg * 16 + i) * 128 + c] = celu13(accv[i] + cb);
    __syncthreads();

    int w = t >> 5, lane = t & 31;
    for (int i = 0; i < 4; ++i) {
        int row = w * 4 + i;
        int c4 = lane * 4;
        float x0 = cbuf[row * 128 + c4 + 0];
        float x1 = cbuf[row * 128 + c4 + 1];
        float x2 = cbuf[row * 128 + c4 + 2];
        float x3 = cbuf[row * 128 + c4 + 3];
        float s = x0 + x1 + x2 + x3;
        float q = x0 * x0 + x1 * x1 + x2 * x2 + x3 * x3;
        #pragma unroll
        for (int m = 1; m <= 16; m <<= 1) {
            s += __shfl_xor(s, m);
            q += __shfl_xor(q, m);
        }
        float mu = s * (1.f / 128.f);
        float var = q * (1.f / 128.f) - mu * mu;
        float rs = rsqrtf(var + 1e-5f);
        #pragma unroll
        for (int cc = 0; cc < 4; ++cc) {
            int col = c4 + cc;
            float x = cbuf[row * 128 + col];
            qout[row * EDIM + g * 128 + col] =
                (x - mu) * rs * gwv[g * 128 + col] + gbv[g * 128 + col];
        }
    }
}

// ============================================================
// Fused projection GEMM (bf16 WMMA, async double-buffered) + CELU + GroupNorm
//   MODE 1 (v-pass): store normalized result as bf16 to vout
//   MODE 0 (k-pass): per-head SCAtt MLP via WMMA -> logits + pool atomics
// ============================================================
template <int MODE>
__global__ __launch_bounds__(256)
void fused_proj_kernel(const float* __restrict__ X,             // [B*M, E]
                       const unsigned short* __restrict__ WT,   // [n][k] bf16
                       const float* __restrict__ bvec,
                       const float* __restrict__ gwv,
                       const float* __restrict__ gbv,
                       const float* __restrict__ q1f,           // MODE 0
                       const unsigned short* __restrict__ WmT,  // [64][128] bf16
                       const float* __restrict__ bmv,
                       const float* __restrict__ Wsv,
                       const float* __restrict__ bsv,
                       float* __restrict__ logits,
                       float* __restrict__ poolg,
                       unsigned short* __restrict__ vout)       // MODE 1
{
    // double-buffered staging; C tile overlays the same region after the GEMM
    __shared__ unsigned short smem[2 * TM * KPAD + 2 * TN * KPAD];
    __shared__ float bias_s[TN], gw_s[TN], gb_s[TN];
    __shared__ float q1s[TN];
    __shared__ float pool_s[128];
    __shared__ float bm_s[MID], ws_s[MID];

    const int t = threadIdx.x;
    const int lane = t & 31;
    const int w = t >> 5;
    const int row0 = blockIdx.x * TM;
    const int b = row0 >> 11;
    const int n_blk = blockIdx.y * TN;

    bias_s[t] = bvec[n_blk + t];
    gw_s[t]   = gwv[n_blk + t];
    gb_s[t]   = gbv[n_blk + t];
    if (MODE == 0) {
        q1s[t] = q1f[b * EDIM + n_blk + t];
        if (t < MID) { bm_s[t] = bmv[t]; ws_s[t] = Wsv[t]; }
        if (t < 128) pool_s[t] = 0.f;
    }

    unsigned short* AsB = smem;                     // [2][TM*KPAD]
    unsigned short* BtB = smem + 2 * TM * KPAD;     // [2][TN*KPAD]

    const int m0 = (w >> 1) * 16;
    const int n0 = (w & 1) * 128;
    const int khA = lane >> 4;
    const int mlA = m0 + (lane & 15);

    v8f acc[8];
    #pragma unroll
    for (int j = 0; j < 8; ++j)
        #pragma unroll
        for (int e = 0; e < 8; ++e) acc[j][e] = 0.f;

    const int arow = t >> 2;
    const int acol = (t & 3) * 8;
    const float* Abase = X + (row0 + arow) * EDIM + acol;
    const unsigned short* Bbase = WT + (n_blk + t) * EDIM;
    unsigned short* myA0 = AsB + arow * KPAD + acol;
    const unsigned btlds = (unsigned)(unsigned long)(BtB + t * KPAD);

    // ---- pipeline stage 0 ----
    {
        async_b128(btlds, (const void*)Bbase, 0);
        async_b128(btlds, (const void*)Bbase, 16);
        async_b128(btlds, (const void*)Bbase, 32);
        async_b128(btlds, (const void*)Bbase, 48);
        f32x4 a0 = *(const f32x4*)(Abase);
        f32x4 a1 = *(const f32x4*)(Abase + 4);
        u32x4 pa;
        pa.x = (unsigned)f2bf(a0.x) | ((unsigned)f2bf(a0.y) << 16);
        pa.y = (unsigned)f2bf(a0.z) | ((unsigned)f2bf(a0.w) << 16);
        pa.z = (unsigned)f2bf(a1.x) | ((unsigned)f2bf(a1.y) << 16);
        pa.w = (unsigned)f2bf(a1.z) | ((unsigned)f2bf(a1.w) << 16);
        *(u32x4*)myA0 = pa;
        wait_async0();
    }
    __syncthreads();

    for (int kb = 0; kb < EDIM / TK; ++kb) {
        const int buf = kb & 1;
        f32x4 na0, na1;
        if (kb < EDIM / TK - 1) {
            // prefetch next K-chunk: B via async DMA, A via registers
            const unsigned short* gsrc = Bbase + (kb + 1) * TK;
            unsigned lb = btlds + (unsigned)((buf ^ 1) * TN * KPAD * 2 - buf * 0);
            lb = (unsigned)(unsigned long)(BtB + (buf ^ 1) * TN * KPAD + t * KPAD);
            async_b128(lb, (const void*)gsrc, 0);
            async_b128(lb, (const void*)gsrc, 16);
            async_b128(lb, (const void*)gsrc, 32);
            async_b128(lb, (const void*)gsrc, 48);
            na0 = *(const f32x4*)(Abase + (kb + 1) * TK);
            na1 = *(const f32x4*)(Abase + (kb + 1) * TK + 4);
        }
        // ---- compute current buffer: hoisted fragment loads, then 8 WMMAs ----
        const unsigned short* Asb = AsB + buf * TM * KPAD;
        const unsigned short* Btb = BtB + buf * TN * KPAD;
        FragU af;
        af.u[0] = *(const u32x4*)&Asb[mlA * KPAD + khA * 8];
        af.u[1] = *(const u32x4*)&Asb[mlA * KPAD + 16 + khA * 8];
        FragU bfr[8];
        #pragma unroll
        for (int j = 0; j < 8; ++j) {
            const u32x4* q = (const u32x4*)&Btb[(n0 + j * 16 + (lane & 15)) * KPAD + khA * 8];
            bfr[j].u[0] = q[0];
            bfr[j].u[1] = q[2];
        }
        #pragma unroll
        for (int j = 0; j < 8; ++j)
            acc[j] = __builtin_amdgcn_wmma_f32_16x16x32_bf16(
                false, af.v, false, bfr[j].v, (short)0, acc[j], false, false);

        if (kb < EDIM / TK - 1) {
            u32x4 pa;
            pa.x = (unsigned)f2bf(na0.x) | ((unsigned)f2bf(na0.y) << 16);
            pa.y = (unsigned)f2bf(na0.z) | ((unsigned)f2bf(na0.w) << 16);
            pa.z = (unsigned)f2bf(na1.x) | ((unsigned)f2bf(na1.y) << 16);
            pa.w = (unsigned)f2bf(na1.z) | ((unsigned)f2bf(na1.w) << 16);
            *(u32x4*)(AsB + (buf ^ 1) * TM * KPAD + arow * KPAD + acol) = pa;
            wait_async0();
        }
        __syncthreads();
    }

    // ---- epilogue: accum + bias -> CELU -> C (bf16 in LDS) ----
    unsigned short* C = smem;   // [64][CPAD]
    #pragma unroll
    for (int j = 0; j < 8; ++j) {
        int n = n0 + j * 16 + (lane & 15);
        float cb = bias_s[n];
        #pragma unroll
        for (int r = 0; r < 8; ++r) {
            int m = m0 + r + 8 * (lane >> 4);
            C[m * CPAD + n] = f2bf(celu13(acc[j][r] + cb));
        }
    }
    __syncthreads();

    // ---- GroupNorm: one wave per (row, head) pair, 16 pairs/wave ----
    for (int i = 0; i < 16; ++i) {
        int p = w * 16 + i;
        int row = p >> 1, lh = p & 1;
        int cb4 = lh * 128 + lane * 4;
        float x0 = bf2f(C[row * CPAD + cb4 + 0]);
        float x1 = bf2f(C[row * CPAD + cb4 + 1]);
        float x2 = bf2f(C[row * CPAD + cb4 + 2]);
        float x3 = bf2f(C[row * CPAD + cb4 + 3]);
        float s = x0 + x1 + x2 + x3;
        float q = x0 * x0 + x1 * x1 + x2 * x2 + x3 * x3;
        #pragma unroll
        for (int m = 1; m <= 16; m <<= 1) {
            s += __shfl_xor(s, m);
            q += __shfl_xor(q, m);
        }
        float mu = s * (1.f / 128.f);
        float var = q * (1.f / 128.f) - mu * mu;
        float rs = rsqrtf(var + 1e-5f);
        float xs[4] = {x0, x1, x2, x3};
        #pragma unroll
        for (int cc = 0; cc < 4; ++cc) {
            int col = cb4 + cc;
            C[row * CPAD + col] = f2bf((xs[cc] - mu) * rs * gw_s[col] + gb_s[col]);
        }
    }
    __syncthreads();

    if (MODE == 1) {
        int row = t >> 2;
        int cc0 = (t & 3) * 64;
        #pragma unroll
        for (int k8 = 0; k8 < 8; ++k8) {
            u32x4 d = *(const u32x4*)&C[row * CPAD + cc0 + k8 * 8];
            *(u32x4*)(vout + (row0 + row) * EDIM + n_blk + cc0 + k8 * 8) = d;
        }
    } else {
        // ---- SCAtt MLP via WMMA: A = (q1 ⊙ k̂)[64x128], B = Wm[128x64] per head ----
        int lh = w >> 2;
        int mm0 = (w & 3) * 16;
        v8f macc[4];
        #pragma unroll
        for (int j = 0; j < 4; ++j)
            #pragma unroll
            for (int e = 0; e < 8; ++e) macc[j][e] = 0.f;

        const int kh = lane >> 4;
        const int ml = mm0 + (lane & 15);
        #pragma unroll
        for (int ks = 0; ks < 4; ++ks) {
            FragU af;
            int d0 = ks * 32 + kh * 8;
            int d1 = ks * 32 + 16 + kh * 8;
            #pragma unroll
            for (int vv = 0; vv < 8; ++vv) {
                int da = d0 + vv, db = d1 + vv;
                af.s[vv]     = f2bf(bf2f(C[ml * CPAD + lh * 128 + da]) * q1s[lh * 128 + da]);
                af.s[8 + vv] = f2bf(bf2f(C[ml * CPAD + lh * 128 + db]) * q1s[lh * 128 + db]);
            }
            #pragma unroll
            for (int j = 0; j < 4; ++j) {
                FragU bf_;
                const u32x4* q =
                    (const u32x4*)(WmT + (j * 16 + (lane & 15)) * 128 + ks * 32 + kh * 8);
                bf_.u[0] = q[0];
                bf_.u[1] = q[2];
                macc[j] = __builtin_amdgcn_wmma_f32_16x16x32_bf16(
                    false, af.v, false, bf_.v, (short)0, macc[j], false, false);
            }
        }
        float lsum[8];
        #pragma unroll
        for (int r = 0; r < 8; ++r) lsum[r] = 0.f;
        #pragma unroll
        for (int j = 0; j < 4; ++j) {
            int n = j * 16 + (lane & 15);
            float bmn = bm_s[n], wsn = ws_s[n];
            float ps = 0.f;
            #pragma unroll
            for (int r = 0; r < 8; ++r) {
                float vv = fmaxf(macc[j][r] + bmn, 0.f);
                lsum[r] = fmaf(vv, wsn, lsum[r]);
                ps += vv;
            }
            atomicAdd(&pool_s[lh * 64 + n], ps);
        }
        #pragma unroll
        for (int m = 1; m <= 8; m <<= 1)
            #pragma unroll
            for (int r = 0; r < 8; ++r) lsum[r] += __shfl_xor(lsum[r], m);
        if ((lane & 15) == 0) {
            int h = blockIdx.y * 2 + lh;
            float bsc = bsv[0];
            #pragma unroll
            for (int r = 0; r < 8; ++r) {
                int mrow = row0 - b * MLEN + mm0 + r + 8 * kh;
                logits[(b * HEADS + h) * MLEN + mrow] = lsum[r] + bsc;
            }
        }
        __syncthreads();
        if (t < 128) {
            int lh2 = t >> 6, n = t & 63;
            atomicAdd(&poolg[(b * HEADS + blockIdx.y * 2 + lh2) * MID + n], pool_s[t]);
        }
    }
}

// ============================================================
// Finalize per (b,h): channel gate, softmax over M, weighted v-agg, combine q2
// ============================================================
__global__ __launch_bounds__(256)
void finalize_kernel(const unsigned short* __restrict__ vbf,
                     const float* __restrict__ logits,
                     const float* __restrict__ poolg,
                     const float* __restrict__ q2f,
                     const float* __restrict__ Wc, const float* __restrict__ bc,
                     float* __restrict__ out) {
    __shared__ float wl[MLEN];
    __shared__ float red[256];
    __shared__ float pl[MID];
    __shared__ float ach[HD];
    __shared__ float part0[256], part1[256];
    const int bh = blockIdx.x, b = bh >> 3, h = bh & 7;
    const int t = threadIdx.x;
    const float* lg = logits + bh * MLEN;

    float l8[8], mx = -3.4e38f;
    #pragma unroll
    for (int i = 0; i < 8; ++i) { l8[i] = lg[t * 8 + i]; mx = fmaxf(mx, l8[i]); }
    red[t] = mx;
    __syncthreads();
    for (int s2 = 128; s2 > 0; s2 >>= 1) {
        if (t < s2) red[t] = fmaxf(red[t], red[t + s2]);
        __syncthreads();
    }
    mx = red[0];
    __syncthreads();
    float sm = 0.f;
    #pragma unroll
    for (int i = 0; i < 8; ++i) {
        float e = expf(l8[i] - mx);
        wl[t * 8 + i] = e;
        sm += e;
    }
    red[t] = sm;
    __syncthreads();
    for (int s2 = 128; s2 > 0; s2 >>= 1) {
        if (t < s2) red[t] += red[t + s2];
        __syncthreads();
    }
    float inv = 1.f / red[0];

    if (t < MID) pl[t] = poolg[bh * MID + t] * (1.f / (float)MLEN);
    __syncthreads();
    if (t < HD) {
        float a = bc[t];
        for (int mid = 0; mid < MID; ++mid) a = fmaf(pl[mid], Wc[mid * HD + t], a);
        ach[t] = 1.f / (1.f + expf(-a));
    }

    // spatially weighted value aggregation: u32 loads carry bf16 pairs,
    // M split 4 ways across thread groups for wider per-wave requests.
    const int c = t & 63, quarter = t >> 6;
    const unsigned* vb32 =
        (const unsigned*)vbf + (size_t)(b * MLEN + quarter * 512) * (EDIM / 2) + h * 64 + c;
    float a0 = 0.f, a1 = 0.f;
    for (int m = 0; m < 512; ++m) {
        unsigned u = vb32[(size_t)m * (EDIM / 2)];
        float wgt = wl[quarter * 512 + m];
        a0 = fmaf(wgt, bf2f((unsigned short)(u & 0xFFFFu)), a0);
        a1 = fmaf(wgt, bf2f((unsigned short)(u >> 16)), a1);
    }
    part0[t] = a0;
    part1[t] = a1;
    __syncthreads();
    if (t < 64) {
        float s0 = part0[t] + part0[t + 64] + part0[t + 128] + part0[t + 192];
        float s1 = part1[t] + part1[t + 64] + part1[t + 128] + part1[t + 192];
        int d0 = 2 * t, d1 = 2 * t + 1;
        out[b * EDIM + h * HD + d0] = q2f[b * EDIM + h * HD + d0] * (s0 * inv) * ach[d0];
        out[b * EDIM + h * HD + d1] = q2f[b * EDIM + h * HD + d1] * (s1 * inv) * ach[d1];
    }
}

// ============================================================
// launch
// ============================================================
extern "C" void kernel_launch(void* const* d_in, const int* in_sizes, int n_in,
                              void* d_out, int out_size, void* d_ws, size_t ws_size,
                              hipStream_t stream) {
    (void)in_sizes; (void)n_in; (void)out_size; (void)ws_size;
    const float* query = (const float*)d_in[0];
    const float* key   = (const float*)d_in[1];
    const float* value = (const float*)d_in[2];
    const float* Wq  = (const float*)d_in[3];
    const float* bq  = (const float*)d_in[4];
    const float* gqw = (const float*)d_in[5];
    const float* gqb = (const float*)d_in[6];
    const float* Wv1  = (const float*)d_in[7];
    const float* bv1  = (const float*)d_in[8];
    const float* gv1w = (const float*)d_in[9];
    const float* gv1b = (const float*)d_in[10];
    const float* Wk  = (const float*)d_in[11];
    const float* bk  = (const float*)d_in[12];
    const float* gkw = (const float*)d_in[13];
    const float* gkb = (const float*)d_in[14];
    const float* Wv2  = (const float*)d_in[15];
    const float* bv2  = (const float*)d_in[16];
    const float* gv2w = (const float*)d_in[17];
    const float* gv2b = (const float*)d_in[18];
    const float* Wm = (const float*)d_in[19];
    const float* bm = (const float*)d_in[20];
    const float* Ws = (const float*)d_in[21];
    const float* bs = (const float*)d_in[22];
    const float* Wc = (const float*)d_in[23];
    const float* bc = (const float*)d_in[24];

    // workspace layout
    char* ws = (char*)d_ws;
    unsigned short* WkT  = (unsigned short*)(ws + 0);                    // 2 MB
    unsigned short* WvT  = (unsigned short*)(ws + (2u << 20));           // 2 MB
    unsigned short* WmT  = (unsigned short*)(ws + (4u << 20));           // 16 KB
    float* q1f    = (float*)(ws + (4u << 20) + 16384);                   // 128 KB
    float* q2f    = (float*)(ws + (4u << 20) + 16384 + 131072);          // 128 KB
    float* logits = (float*)(ws + (4u << 20) + 16384 + 262144);          // 2 MB
    float* pool   = (float*)(ws + (6u << 20) + 16384 + 262144);          // 64 KB
    unsigned short* vbf = (unsigned short*)(ws + (6u << 20) + 16384 + 262144 + 65536); // 128 MB

    transpose_to_bf16<<<dim3(32, 32), dim3(32, 8), 0, stream>>>(Wk, WkT, EDIM, EDIM);
    transpose_to_bf16<<<dim3(32, 32), dim3(32, 8), 0, stream>>>(Wv2, WvT, EDIM, EDIM);
    transpose_to_bf16<<<dim3(2, 4),  dim3(32, 8), 0, stream>>>(Wm, WmT, HD, MID);
    zero_kernel<<<64, 256, 0, stream>>>(pool, BATCH * HEADS * MID);

    qproj_kernel<<<HEADS, 256, 0, stream>>>(query, Wq,  bq,  gqw,  gqb,  q1f);
    qproj_kernel<<<HEADS, 256, 0, stream>>>(query, Wv1, bv1, gv1w, gv1b, q2f);

    fused_proj_kernel<1><<<dim3(BATCH * MLEN / TM, EDIM / TN), 256, 0, stream>>>(
        value, WvT, bv2, gv2w, gv2b,
        nullptr, nullptr, nullptr, nullptr, nullptr, nullptr, nullptr, vbf);

    fused_proj_kernel<0><<<dim3(BATCH * MLEN / TM, EDIM / TN), 256, 0, stream>>>(
        key, WkT, bk, gkw, gkb,
        q1f, WmT, bm, Ws, bs, logits, pool, nullptr);

    finalize_kernel<<<BATCH * HEADS, 256, 0, stream>>>(
        vbf, logits, pool, q2f, Wc, bc, (float*)d_out);
}
